// gabor_fixedResponse_3882650435768
// MI455X (gfx1250) — matrix operand order repaired and checked by reading, not compiled, not run on hardware
//
#include <hip/hip_runtime.h>
#include <hip/hip_bf16.h>
#include <math.h>

// ---------------------------------------------------------------------------
// Gabor filterbank generation for MI455X (gfx1250, wave32, WMMA).
//
// out[b,f,t] = flip_t( env(b,f)*carrier(f) ) / maxfft[f]
// maxfft[f]  = max_{b,k} | rfft_512( y[b,f,0:512] )[k] |
//
// Strategy:
//   kernel 1: build f16 DFT basis (cos/sin, bin-major) into d_ws (lives in L2)
//   kernel 2: one workgroup per filter f (512 WGs, 256 thr = 8 wave32):
//     A) compute y[b, 0:512] as f16 into LDS (stride 520 halves: bank-clean)
//     B) DFT-as-GEMM with v_wmma_f32_16x16x32_f16: M=64 batches, K=512,
//        N=272 bins (257 real + zero pad).  max |X|^2 -> LDS atomicMax.
//     C) recompute y (full T=1024), scale by rsqrt(max|X|^2), write flipped.
// ---------------------------------------------------------------------------

typedef _Float16 half8  __attribute__((ext_vector_type(8)));
typedef _Float16 half16 __attribute__((ext_vector_type(16)));
typedef float    f32x8  __attribute__((ext_vector_type(8)));

#define N_FILT   512
#define BATCH    64
#define T_OUT    1024
#define FFT_N    512
#define NBIN     257
#define NBIN_PAD 272            // 17 tiles of 16 columns
#define Y_STRIDE 520            // 512 + 8 halves: 4-dword shift/row -> 64 banks
#define FS_F     16000.0f
#define TWO_PI_F 6.2831853071795864f

// ---------------------------------------------------------------------------
// Kernel 1: f16 DFT basis, bin-major.  ws[bin*512 + j] = cos(2*pi*bin*j/512),
// second plane (+NBIN_PAD*FFT_N) = sin(...).  Bins >= 257 are zero padding.
// ---------------------------------------------------------------------------
__global__ void gabor_basis_kernel(_Float16* __restrict__ ws) {
    int idx = blockIdx.x * blockDim.x + threadIdx.x;
    if (idx >= NBIN_PAD * FFT_N) return;
    int bin = idx >> 9;
    int j   = idx & (FFT_N - 1);
    float ph = (TWO_PI_F / (float)FFT_N) * (float)bin * (float)j;
    float s, c;
    __sincosf(ph, &s, &c);
    if (bin >= NBIN) { c = 0.0f; s = 0.0f; }
    ws[idx]                     = (_Float16)c;
    ws[NBIN_PAD * FFT_N + idx]  = (_Float16)s;
}

// ---------------------------------------------------------------------------
// Kernel 2: per-filter workgroup.
// ---------------------------------------------------------------------------
__launch_bounds__(256, 1)
__global__ void gabor_wmma_kernel(const float* __restrict__ Q,
                                  const float* __restrict__ fc,
                                  float* __restrict__ out,
                                  const _Float16* __restrict__ basis) {
    const int f    = blockIdx.x;        // 0..511
    const int tid  = threadIdx.x;       // 0..255
    const int lane = tid & 31;
    const int wave = tid >> 5;          // 0..7

    __shared__ _Float16 yld[BATCH * Y_STRIDE];   // 66560 B
    __shared__ float    bq[BATCH];
    __shared__ unsigned maxbits;

    // Token L2 warm-up of the DFT basis (global_prefetch_b8 path).
    __builtin_prefetch(basis + (size_t)tid * 2176, 0, 1);

    const float fcv   = fc[f];
    const float omega = (TWO_PI_F / FS_F) * fcv;
    const float acoef = 0.5f * 2.5066282746310002f;   // 0.5*sqrt(2*pi)

    if (tid < BATCH) {
        float q = Q[tid * N_FILT + f];
        bq[tid] = (fcv / q) * acoef / FS_F;           // b in reference
    }
    if (tid == 0) maxbits = 0u;
    __syncthreads();

    // ---- Phase A: y[b][j], j=0..511 (n = j+1), f16 into LDS -----------------
    for (int idx = tid; idx < BATCH * FFT_N; idx += 256) {
        int   b = idx >> 9;
        int   j = idx & (FFT_N - 1);
        float n = (float)(j + 1);
        float g = bq[b] * n;
        float v = __expf(-g * g) * __cosf(omega * n);
        yld[b * Y_STRIDE + j] = (_Float16)v;
    }
    __syncthreads();

    // ---- Phase B: DFT via WMMA ---------------------------------------------
    // wave -> row tile (batches rt*16..rt*16+15) and a strided set of bin tiles
    const int rt      = wave & 3;
    const int ctStart = wave >> 2;

    // A fragment (16x32 f16): lane<16 -> M=lane,    K chunks {0..7, 16..23}
    //                         lane>=16 -> M=lane-16, K chunks {8..15, 24..31}
    const int Am  = lane & 15;
    const int AkA = (lane < 16) ? 0 : 8;
    const int AkB = AkA + 16;
    const _Float16* yrow = &yld[(rt * 16 + Am) * Y_STRIDE];

    // B fragment (32x16 f16), basis stored bin-major:
    //   lane<16 -> N=lane,    K chunks {0..7, 16..23}; lane>=16 -> N=lane-16, K chunks {8..15, 24..31}
    const int Bn = lane & 15;
    const int Bk = (lane < 16) ? 0 : 8;

    float m = 0.0f;
    for (int ct = ctStart; ct < 17; ct += 2) {          // wave-uniform trip count
        f32x8 accC = {};
        f32x8 accS = {};
        const _Float16* bCrow = basis + (size_t)(ct * 16 + Bn) * FFT_N + Bk;
        const _Float16* bSrow = bCrow + (size_t)NBIN_PAD * FFT_N;
        for (int ks = 0; ks < 16; ++ks) {
            const int k0 = ks * 32;
            half8 a_lo = *(const half8*)(yrow + k0 + AkA);   // ds_load_b128
            half8 a_hi = *(const half8*)(yrow + k0 + AkB);   // ds_load_b128
            half16 a = __builtin_shufflevector(a_lo, a_hi,
                        0,1,2,3,4,5,6,7,8,9,10,11,12,13,14,15);
            half8 c_lo = *(const half8*)(bCrow + k0);
            half8 c_hi = *(const half8*)(bCrow + k0 + 16);
            half16 bc = __builtin_shufflevector(c_lo, c_hi,
                        0,1,2,3,4,5,6,7,8,9,10,11,12,13,14,15);
            half8 s_lo = *(const half8*)(bSrow + k0);
            half8 s_hi = *(const half8*)(bSrow + k0 + 16);
            half16 bs = __builtin_shufflevector(s_lo, s_hi,
                        0,1,2,3,4,5,6,7,8,9,10,11,12,13,14,15);
            accC = __builtin_amdgcn_wmma_f32_16x16x32_f16(
                       false, a, false, bc, (short)0, accC, false, false);
            accS = __builtin_amdgcn_wmma_f32_16x16x32_f16(
                       false, a, false, bs, (short)0, accS, false, false);
        }
#pragma unroll
        for (int i = 0; i < 8; ++i) {
            float mag2 = accC[i] * accC[i] + accS[i] * accS[i];
            m = fmaxf(m, mag2);
        }
    }
    // Positive floats compare correctly as unsigned bit patterns.
    atomicMax(&maxbits, __float_as_uint(m));
    __syncthreads();

    // filters = y / maxfft ; maxfft = sqrt(max |X|^2)  ->  multiply by rsqrt
    const float inv = __frsqrt_rn(__uint_as_float(maxbits));

    // ---- Phase C: recompute full y (T=1024), scale, write flipped ----------
    for (int idx = tid; idx < BATCH * T_OUT; idx += 256) {
        int   b = idx >> 10;
        int   t = idx & (T_OUT - 1);
        float n = (float)(T_OUT - t);     // flip: out[t] <- y[T-1-t], n = j+1
        float g = bq[b] * n;
        float v = __expf(-g * g) * __cosf(omega * n) * inv;
        out[(((size_t)b * N_FILT) + f) * T_OUT + t] = v;
    }
}

// ---------------------------------------------------------------------------
extern "C" void kernel_launch(void* const* d_in, const int* in_sizes, int n_in,
                              void* d_out, int out_size, void* d_ws, size_t ws_size,
                              hipStream_t stream) {
    (void)in_sizes; (void)n_in; (void)out_size; (void)ws_size;
    const float* Q  = (const float*)d_in[0];   // [64, 512]
    const float* fc = (const float*)d_in[1];   // [512]
    float* out      = (float*)d_out;           // [64, 512, 1024]
    _Float16* basis = (_Float16*)d_ws;         // 2 * 272*512 f16 = 557056 B

    const int nElem  = NBIN_PAD * FFT_N;
    const int nBlock = (nElem + 255) / 256;
    gabor_basis_kernel<<<nBlock, 256, 0, stream>>>(basis);
    gabor_wmma_kernel<<<N_FILT, 256, 0, stream>>>(Q, fc, out, basis);
}
